// CoreAttention_8375186227230
// MI455X (gfx1250) — matrix operand order repaired and checked by reading, not compiled, not run on hardware
//
#include <hip/hip_runtime.h>

typedef __attribute__((ext_vector_type(16))) _Float16 v16h;
typedef __attribute__((ext_vector_type(8)))  _Float16 v8h;
typedef __attribute__((ext_vector_type(8)))  float    v8f;
typedef __attribute__((ext_vector_type(4)))  float    v4f;

#define SEQ   2048
#define HDIM  64
#define WAVES 8
#define QROWS 32                 // query rows per wave (2 x 16-row WMMA A tiles)
#define QT_PER_BH (SEQ / QROWS)  // 64 query tiles per (b,h)
#define BH_TOTAL 32              // B*H = 2*16

__device__ __forceinline__ float red_max16(float v) {
    #pragma unroll
    for (int off = 8; off >= 1; off >>= 1)
        v = fmaxf(v, __shfl_xor(v, off, 32));
    return v;
}
__device__ __forceinline__ float red_add16(float v) {
    #pragma unroll
    for (int off = 8; off >= 1; off >>= 1)
        v += __shfl_xor(v, off, 32);
    return v;
}

__global__ __launch_bounds__(WAVES * 32, 1)
void fa_causal_wmma(const float* __restrict__ Q, const float* __restrict__ K,
                    const float* __restrict__ V, float* __restrict__ O)
{
    // per-wave P staging tile (C-layout -> A-layout transpose through LDS)
    __shared__ __align__(32) _Float16 ldsP[WAVES][16][32];

    const int tid  = threadIdx.x;
    const int wave = tid >> 5;
    const int lane = tid & 31;
    const int llo  = lane & 15;   // column / row-in-half index
    const int lhi  = lane >> 4;   // which 16-lane half

    const int gw    = blockIdx.x * WAVES + wave;
    const int bh    = gw / QT_PER_BH;
    const int qbase = (gw % QT_PER_BH) * QROWS;   // 32-aligned

    const float* Qb = Q + (size_t)bh * SEQ * HDIM;
    const float* Kb = K + (size_t)bh * SEQ * HDIM;
    const float* Vb = V + (size_t)bh * SEQ * HDIM;
    float*       Ob = O + (size_t)bh * SEQ * HDIM;

    // ---- Q tiles as WMMA A operands, pre-scaled by 1/sqrt(64) ----
    // A operand, contiguous k order chosen here: lane row = llo; elem e -> K = 16*lhi + e
    v16h qa[2][2];   // [q subtile][d slice]
    #pragma unroll
    for (int qt = 0; qt < 2; ++qt) {
        const float* qp = Qb + (size_t)(qbase + qt * 16 + llo) * HDIM;
        #pragma unroll
        for (int s = 0; s < 2; ++s) {
            v4f f0 = *(const v4f*)(qp + s * 32 + 16 * lhi);
            v4f f1 = *(const v4f*)(qp + s * 32 + 16 * lhi + 4);
            v4f f2 = *(const v4f*)(qp + s * 32 + 16 * lhi + 8);
            v4f f3 = *(const v4f*)(qp + s * 32 + 16 * lhi + 12);
            #pragma unroll
            for (int e = 0; e < 4; ++e) {
                qa[qt][s][e]      = (_Float16)(f0[e] * 0.125f);
                qa[qt][s][e + 4]  = (_Float16)(f1[e] * 0.125f);
                qa[qt][s][e + 8]  = (_Float16)(f2[e] * 0.125f);
                qa[qt][s][e + 12] = (_Float16)(f3[e] * 0.125f);
            }
        }
    }

    const v8f vzero = {};
    v8f oacc[2][4] = {{vzero, vzero, vzero, vzero}, {vzero, vzero, vzero, vzero}};
    float mrow[2][8], lrow[2][8];      // row stats: row = qbase + 16*qt + i + 8*lhi
    #pragma unroll
    for (int qt = 0; qt < 2; ++qt)
        #pragma unroll
        for (int i = 0; i < 8; ++i) { mrow[qt][i] = -3.0e38f; lrow[qt][i] = 0.0f; }

    const int kvEnd = qbase + QROWS;   // causal: kv0 runs 0..qbase (32-aligned)
    for (int kv0 = 0; kv0 < kvEnd; kv0 += 32) {
        // ============ phase 1: scores for both subtiles (kb transient) ============
        v8f c[2][2];     // [q subtile][kv half]
        {
            // B operand: lane col = llo (kv row kv0+16h+llo), K dim = 32s + 16*lhi + e
            v16h kb[2][2];   // [kv half h][d slice s]
            #pragma unroll
            for (int h = 0; h < 2; ++h) {
                const float* kp = Kb + (size_t)(kv0 + h * 16 + llo) * HDIM + lhi * 16;
                #pragma unroll
                for (int q4 = 0; q4 < 4; ++q4) {
                    v4f a = *(const v4f*)(kp + q4 * 4);
                    v4f b = *(const v4f*)(kp + 32 + q4 * 4);
                    #pragma unroll
                    for (int e = 0; e < 4; ++e) {
                        kb[h][0][q4 * 4 + e] = (_Float16)a[e];
                        kb[h][1][q4 * 4 + e] = (_Float16)b[e];
                    }
                }
            }
            #pragma unroll
            for (int qt = 0; qt < 2; ++qt) {
                #pragma unroll
                for (int h = 0; h < 2; ++h) {
                    v8f acc = {};
                    acc = __builtin_amdgcn_wmma_f32_16x16x32_f16(false, qa[qt][0], false, kb[h][0],
                                                                 (short)0, acc, false, false);
                    acc = __builtin_amdgcn_wmma_f32_16x16x32_f16(false, qa[qt][1], false, kb[h][1],
                                                                 (short)0, acc, false, false);
                    c[qt][h] = acc;
                }
            }
        }

        // ============ phase 2: mask + online softmax -> pa[2] (c dies) ============
        v16h pa[2];
        #pragma unroll
        for (int qt = 0; qt < 2; ++qt) {
            const int qsub = qbase + qt * 16;

            // causal mask: only the diagonal kv tile needs it (kv0 <= qbase)
            if (kv0 + 31 > qsub) {
                #pragma unroll
                for (int i = 0; i < 8; ++i) {
                    const int qrow = qsub + i + 8 * lhi;
                    if (kv0 + llo      > qrow) c[qt][0][i] = -10000.0f;
                    if (kv0 + 16 + llo > qrow) c[qt][1][i] = -10000.0f;
                }
            }

            #pragma unroll
            for (int i = 0; i < 8; ++i) {
                const float mt    = red_max16(fmaxf(c[qt][0][i], c[qt][1][i]));
                const float mnew  = fmaxf(mrow[qt][i], mt);
                const float alpha = __expf(mrow[qt][i] - mnew);
                mrow[qt][i] = mnew;
                const float p0 = __expf(c[qt][0][i] - mnew);
                const float p1 = __expf(c[qt][1][i] - mnew);
                lrow[qt][i] = lrow[qt][i] * alpha + red_add16(p0 + p1);
                #pragma unroll
                for (int t = 0; t < 4; ++t) oacc[qt][t][i] *= alpha;
                const int pr = i + 8 * lhi;
                ldsP[wave][pr][llo]      = (_Float16)p0;  // C layout -> LDS row-major
                ldsP[wave][pr][llo + 16] = (_Float16)p1;
            }

            // reload P in A layout (same-wave DS ops are in-order; no barrier)
            v8h plo = *(const v8h*)&ldsP[wave][llo][16 * lhi];
            v8h phi = *(const v8h*)&ldsP[wave][llo][16 * lhi + 8];
            pa[qt] = __builtin_shufflevector(plo, phi,
                                             0, 1, 2, 3, 4, 5, 6, 7,
                                             8, 9, 10, 11, 12, 13, 14, 15);
        }

        // ============ phase 3: O += P * V, one V column-tile at a time ============
        #pragma unroll
        for (int t = 0; t < 4; ++t) {
            // B operand: lane col = output d = t*16 + llo; K dim = kv0 + 16*lhi + e
            const float* vp = Vb + (size_t)(kv0 + 16 * lhi) * HDIM + t * 16 + llo;
            v16h vb;
            #pragma unroll
            for (int e = 0; e < 16; ++e)
                vb[e] = (_Float16)vp[(size_t)e * HDIM];
            #pragma unroll
            for (int qt = 0; qt < 2; ++qt)
                oacc[qt][t] = __builtin_amdgcn_wmma_f32_16x16x32_f16(false, pa[qt], false, vb,
                                                                     (short)0, oacc[qt][t],
                                                                     false, false);
        }

        // prefetch next kv tile's K / V rows into cache
        if (kv0 + 32 < kvEnd) {
            __builtin_prefetch(Kb + (size_t)(kv0 + 32 + llo) * HDIM, 0, 3);
            __builtin_prefetch(Vb + (size_t)(kv0 + 32 + llo) * HDIM, 0, 3);
        }
    }

    // ---------- epilogue: divide by row sums, coalesced fp32 stores ----------
    #pragma unroll
    for (int qt = 0; qt < 2; ++qt) {
        #pragma unroll
        for (int i = 0; i < 8; ++i) {
            const float inv = 1.0f / lrow[qt][i];
            float* op = Ob + (size_t)(qbase + qt * 16 + i + 8 * lhi) * HDIM + llo;
            #pragma unroll
            for (int t = 0; t < 4; ++t)
                op[t * 16] = oacc[qt][t][i] * inv;
        }
    }
}

extern "C" void kernel_launch(void* const* d_in, const int* in_sizes, int n_in,
                              void* d_out, int out_size, void* d_ws, size_t ws_size,
                              hipStream_t stream) {
    (void)in_sizes; (void)n_in; (void)out_size; (void)d_ws; (void)ws_size;
    const float* Q = (const float*)d_in[0];
    const float* K = (const float*)d_in[1];
    const float* V = (const float*)d_in[2];
    float* O = (float*)d_out;

    const int totalWaves = BH_TOTAL * QT_PER_BH;       // 32 * 64 = 2048
    dim3 grid(totalWaves / WAVES);                     // 256 blocks
    dim3 block(WAVES * 32);                            // 256 threads = 8 waves
    fa_causal_wmma<<<grid, block, 0, stream>>>(Q, K, V, O);
}
